// CausalSelfAttention_46634754899970
// MI455X (gfx1250) — matrix operand order repaired and checked by reading, not compiled, not run on hardware
//
#include <hip/hip_runtime.h>
#include <hip/hip_bf16.h>
#include <math.h>
#include <stdint.h>

// Problem constants (match reference)
#define BB 4
#define TT 2048
#define DD 1024
#define HH 16
#define HKV 4
#define HD 64
#define MTOT (BB * TT)          // 8192 rows
#define EPSV 1.1920928955078125e-07f
#define LOG2E 1.44269504088896340736f
#define LOG2_10000 13.287712379549449f

typedef __attribute__((ext_vector_type(16))) _Float16 v16h;
typedef __attribute__((ext_vector_type(8)))  _Float16 v8h;
typedef __attribute__((ext_vector_type(4)))  _Float16 v4h;
typedef __attribute__((ext_vector_type(8)))  float    v8f;

// ---------------------------------------------------------------------------
// CDNA5 async global->LDS copy (ASYNCcnt tracked). lds_off is the byte offset
// within the workgroup LDS allocation (low 32 bits of a flat shared pointer).
// ---------------------------------------------------------------------------
__device__ __forceinline__ void async_load_b128(unsigned lds_off,
                                                const void* gaddr) {
  asm volatile("global_load_async_to_lds_b128 %0, %1, off"
               :: "v"(lds_off), "v"((unsigned long long)(uintptr_t)gaddr)
               : "memory");
}
__device__ __forceinline__ void wait_async0() {
  asm volatile("s_wait_asynccnt 0" ::: "memory");
}
__device__ __forceinline__ unsigned lds_off_of(const void* p) {
  return (unsigned)(uintptr_t)p;
}
// lane <-> lane^16 exchange as pure VALU (v_permlanex16_b32, identity sels)
__device__ __forceinline__ float xor16(float x) {
  int r = __builtin_amdgcn_permlanex16(__float_as_int(x), __float_as_int(x),
                                       0x76543210, 0xfedcba98, false, false);
  return __int_as_float(r);
}
// build a 16x32 f16 WMMA fragment from two 16-byte LDS/global chunks
__device__ __forceinline__ v16h frag16(const _Float16* p0,
                                       const _Float16* p1) {
  v8h lo = *(const v8h*)p0;
  v8h hi = *(const v8h*)p1;
  v16h f;
#pragma unroll
  for (int e = 0; e < 8; ++e) { f[e] = lo[e]; f[8 + e] = hi[e]; }
  return f;
}

// ---------------------------------------------------------------------------
// Tiled GEMM: C[M,N] = A[M,K] * W[N,K]^T   (A: f32 or f16, W: f32, C: f32)
// Block: 256 threads = 8 waves (2M x 4N). Wave tile 32 x (16*NT).
// Block tile 64 x (64*NT). K staged 64 wide, double-buffered LDS.
// All fragments of a burst are loaded before the WMMA burst (partial waits).
// ---------------------------------------------------------------------------
template <typename AT, int NT>
__global__ void gemm_xwt(const AT* __restrict__ A, const float* __restrict__ W,
                         float* __restrict__ C, int M, int N, int K) {
  __shared__ _Float16 As[2][64 * 72];        // [m][k], 144B row stride
  __shared__ _Float16 Bs[2][64 * NT * 72];   // [n][k]
  const int tid  = threadIdx.x;
  const int wave = __builtin_amdgcn_readfirstlane(tid >> 5);  // wave-uniform
  const int lane = tid & 31;
  const int half = lane >> 4;
  const int l16  = lane & 15;
  const int m0 = blockIdx.x * 64;
  const int n0 = blockIdx.y * (64 * NT);
  const int wm = (wave & 1) * 32;
  const int wn = (wave >> 1) * (16 * NT);

  auto stage = [&](int k0, int p) {
    if constexpr (sizeof(AT) == 4) {
#pragma unroll
      for (int it = 0; it < 4; ++it) {
        int idx = tid + it * 256;             // 1024 float4 chunks (64x64)
        int r = idx >> 4, c = (idx & 15) * 4;
        float4 av = *(const float4*)&A[(size_t)(m0 + r) * K + k0 + c];
        v4h hv;
        hv[0] = (_Float16)av.x; hv[1] = (_Float16)av.y;
        hv[2] = (_Float16)av.z; hv[3] = (_Float16)av.w;
        *(v4h*)&As[p][r * 72 + c] = hv;
      }
    } else {
#pragma unroll
      for (int it = 0; it < 2; ++it) {
        int idx = tid + it * 256;             // 512 v8h chunks (64x64 f16)
        int r = idx >> 3, c = (idx & 7) * 8;
        *(v8h*)&As[p][r * 72 + c] =
            *(const v8h*)&A[(size_t)(m0 + r) * K + k0 + c];
      }
    }
#pragma unroll
    for (int it = 0; it < 4 * NT; ++it) {
      int idx = tid + it * 256;
      int r = idx >> 4, c = (idx & 15) * 4;
      float4 wv = *(const float4*)&W[(size_t)(n0 + r) * K + k0 + c];
      v4h hv;
      hv[0] = (_Float16)wv.x; hv[1] = (_Float16)wv.y;
      hv[2] = (_Float16)wv.z; hv[3] = (_Float16)wv.w;
      *(v4h*)&Bs[p][r * 72 + c] = hv;
    }
  };

  v8f acc[2][NT] = {};
  const int S = K / 64;
  stage(0, 0);
  for (int s = 0; s < S; ++s) {
    const int p = s & 1;
    __syncthreads();
    if (s + 1 < S) stage((s + 1) * 64, 1 - p);

#pragma unroll
    for (int k0in = 0; k0in < 64; k0in += 32) {
      // load the whole fragment set first, then burst the WMMAs
      v16h af[2];
#pragma unroll
      for (int i = 0; i < 2; ++i) {
        const _Float16* ap = &As[p][(wm + i * 16 + l16) * 72 + k0in + half * 8];
        af[i] = frag16(ap, ap + 16);
      }
      v16h bfr[NT];
#pragma unroll
      for (int jn = 0; jn < NT; ++jn) {
        const _Float16* bp =
            &Bs[p][(wn + jn * 16 + l16) * 72 + k0in + half * 8];
        bfr[jn] = frag16(bp, bp + 16);
      }
#pragma unroll
      for (int i = 0; i < 2; ++i)
#pragma unroll
        for (int jn = 0; jn < NT; ++jn)
          acc[i][jn] = __builtin_amdgcn_wmma_f32_16x16x32_f16(
              false, af[i], false, bfr[jn], (short)0, acc[i][jn], false, false);
    }
  }

#pragma unroll
  for (int i = 0; i < 2; ++i)
#pragma unroll
    for (int jn = 0; jn < NT; ++jn)
#pragma unroll
      for (int r = 0; r < 8; ++r) {
        int row = m0 + wm + i * 16 + r + half * 8;
        int col = n0 + wn + jn * 16 + l16;
        C[(size_t)row * N + col] = acc[i][jn][r];
      }
}

// ---------------------------------------------------------------------------
// Per-head RMSNorm + RoPE (+ gain*scale*log2e for q). One wave per head row.
// ---------------------------------------------------------------------------
__global__ void normrope_kernel(const float* __restrict__ qf,
                                const float* __restrict__ kf,
                                _Float16* __restrict__ qh,  // [B][H][T][HD]
                                _Float16* __restrict__ kh,  // [B][HKV][T][HD]
                                const float* __restrict__ qgain) {
  const int rid  = blockIdx.x * 8 + (threadIdx.x >> 5);
  const int lane = threadIdx.x & 31;
  const int NQ = BB * TT * HH;

  float x1, x2;
  if (rid < NQ) {
    x1 = qf[(size_t)rid * HD + lane];
    x2 = qf[(size_t)rid * HD + 32 + lane];
  } else {
    int r2 = rid - NQ;
    x1 = kf[(size_t)r2 * HD + lane];
    x2 = kf[(size_t)r2 * HD + 32 + lane];
  }
  float ss = x1 * x1 + x2 * x2;
  ss += __shfl_xor(ss, 1, 32);
  ss += __shfl_xor(ss, 2, 32);
  ss += __shfl_xor(ss, 4, 32);
  ss += __shfl_xor(ss, 8, 32);
  ss += xor16(ss);
  float rms = rsqrtf(ss * (1.0f / 64.0f) + EPSV);
  x1 *= rms;
  x2 *= rms;

  int tpos;
  size_t outbase;
  float gain;
  _Float16* out;
  if (rid < NQ) {
    int h = rid % HH;
    tpos  = (rid / HH) % TT;
    int b = rid / (HH * TT);
    outbase = ((size_t)(b * HH + h) * TT + tpos) * HD;
    gain = qgain[h] * 0.125f * LOG2E;  // fold 1/sqrt(HD) and log2(e) into q
    out = qh;
  } else {
    int r2 = rid - NQ;
    int h = r2 % HKV;
    tpos  = (r2 / HKV) % TT;
    int b = r2 / (HKV * TT);
    outbase = ((size_t)(b * HKV + h) * TT + tpos) * HD;
    gain = 1.0f;
    out = kh;
  }
  // inv = 10000^(-2*lane/64) = exp2(-lane/32 * log2(10000))
  float inv = __builtin_amdgcn_exp2f((float)lane * (-1.0f / 32.0f) * LOG2_10000);
  float ang = (float)tpos * inv;
  float c = __cosf(ang), s = __sinf(ang);
  float y1 = (x1 * c + x2 * s) * gain;
  float y2 = (-x1 * s + x2 * c) * gain;
  out[outbase + lane] = (_Float16)y1;
  out[outbase + 32 + lane] = (_Float16)y2;
}

// ---------------------------------------------------------------------------
// V cast + tile-major transpose. For each 32-token chunk j32, store a
// contiguous 4KB block: vt[((b*HKV+h)*(T/32)+j32)*2048 + hd*32 + (t%32)]
// ---------------------------------------------------------------------------
__global__ void vtrans_kernel(const float* __restrict__ vf,
                              _Float16* __restrict__ vt) {
  const size_t idx = (size_t)blockIdx.x * 256 + threadIdx.x;
  const int tin = (int)(idx & 31);
  const int hd  = (int)((idx >> 5) & 63);
  const int j32 = (int)((idx >> 11) & 63);
  const int h   = (int)((idx >> 17) & 3);
  const int b   = (int)(idx >> 19);
  const int t   = j32 * 32 + tin;
  vt[idx] = (_Float16)vf[((size_t)(b * TT + t)) * (HKV * HD) + h * HD + hd];
}

// ---------------------------------------------------------------------------
// Flash attention. Block = 256 threads = 8 waves = 128 q rows of one (b,h).
// K/V 32-token tiles staged once per block (shared by all 8 waves) with
// double-buffered global_load_async_to_lds_b128; one __syncthreads per chunk.
// Softmax in transposed (A-fragment) domain, base-2 exp, branch-free causal
// masking after exp. Fragment sets loaded before each WMMA burst.
// ---------------------------------------------------------------------------
__global__ void attn_kernel(const _Float16* __restrict__ qh,  // [B][H][T][HD]
                            const _Float16* __restrict__ kh,  // [B][HKV][T][HD]
                            const _Float16* __restrict__ vt,  // tile-major
                            _Float16* __restrict__ yb) {      // [B][T][D]
  __shared__ _Float16 Kbuf[2][32 * 72];  // shared K tile, 144B row stride
  __shared__ _Float16 Vbuf[2][64 * 40];  // shared V tile, 80B row stride
  __shared__ float    Sbuf[8][16 * 36];  // per-wave transpose, 144B stride
  __shared__ float    Abuf[8][16];       // per-wave row-scalar broadcast
  const int tid  = threadIdx.x;
  const int wv   = __builtin_amdgcn_readfirstlane(tid >> 5);  // wave-uniform
  const int lane = tid & 31;
  const int half = lane >> 4;
  const int l16  = lane & 15;

  const int qt = blockIdx.x & 15;   // 16 q-tiles of 128 rows
  const int bh = blockIdx.x >> 4;
  const int b  = bh >> 4;           // H = 16
  const int h  = bh & 15;
  const int kvh = h >> 2;           // GROUPS = 4
  const int qbase = qt * 128 + wv * 16;

  const _Float16* qptr = qh + (((size_t)b * HH + h) * TT + qbase) * HD;
  const _Float16* kcb  = kh + (((size_t)b * HKV + kvh) * TT) * HD;
  const _Float16* vcb  = vt + ((size_t)(b * HKV + kvh) * (TT / 32)) * 2048;

  const unsigned kL[2] = {lds_off_of(&Kbuf[0][0]), lds_off_of(&Kbuf[1][0])};
  const unsigned vL[2] = {lds_off_of(&Vbuf[0][0]), lds_off_of(&Vbuf[1][0])};

  // whole block cooperatively stages one 4KB K tile + one 4KB V tile:
  // 256 threads x 16B = one async round each (1 K + 1 V async op per wave)
  auto issue = [&](int ci, int p) {
    const int j = ci * 32;
    const char* gk = (const char*)(kcb + (size_t)j * HD);
    async_load_b128(kL[p] + (tid >> 3) * 144 + (tid & 7) * 16, gk + tid * 16);
    const char* gv = (const char*)(vcb + (size_t)ci * 2048);
    async_load_b128(vL[p] + (tid >> 2) * 80 + (tid & 3) * 16, gv + tid * 16);
  };

  // Q fragments, held across the whole loop (hd halves [0,32) and [32,64))
  v16h qa[2];
#pragma unroll
  for (int f = 0; f < 2; ++f) {
    const _Float16* base = qptr + (size_t)l16 * HD + f * 32 + half * 8;
    qa[f] = frag16(base, base + 16);
  }

  // softmax state in transposed domain: this lane tracks row l16 (base-2)
  float m_t = -INFINITY, l_t = 0.0f;
  v8f o[4] = {};

  auto process = [&](int j, int p) {
    const _Float16* Kb = &Kbuf[p][0];
    const _Float16* Vb = &Vbuf[p][0];

    // ----- load all 4 K fragments, then burst 4 score WMMAs -----
    v16h kfr[4];
#pragma unroll
    for (int f = 0; f < 2; ++f)
#pragma unroll
      for (int nt = 0; nt < 2; ++nt) {
        const _Float16* kb = Kb + (nt * 16 + l16) * 72 + f * 32 + half * 8;
        kfr[f * 2 + nt] = frag16(kb, kb + 16);
      }
    v8f s[2] = {};
#pragma unroll
    for (int f = 0; f < 2; ++f)
#pragma unroll
      for (int nt = 0; nt < 2; ++nt)
        s[nt] = __builtin_amdgcn_wmma_f32_16x16x32_f16(
            false, qa[f], false, kfr[f * 2 + nt], (short)0, s[nt], false,
            false);

    // ----- LDS round-trip transpose (C layout -> A layout) -----
#pragma unroll
    for (int nt = 0; nt < 2; ++nt)
#pragma unroll
      for (int r = 0; r < 8; ++r)
        Sbuf[wv][(half * 8 + r) * 36 + nt * 16 + l16] = s[nt][r];

    // V fragments issued early so they drain during the softmax math
    v16h vfr[4];
#pragma unroll
    for (int nt2 = 0; nt2 < 4; ++nt2) {
      const _Float16* vb = Vb + (nt2 * 16 + l16) * 40 + half * 8;
      vfr[nt2] = frag16(vb, vb + 16);
    }

    float pv[16];
    {
      const float* sp = &Sbuf[wv][l16 * 36 + half * 8];
#pragma unroll
      for (int e = 0; e < 8; ++e) pv[e] = sp[e];
#pragma unroll
      for (int e = 0; e < 8; ++e) pv[8 + e] = sp[16 + e];
    }

    // ----- online softmax (base-2), tree max over UNMASKED scores -----
    float t8[8];
#pragma unroll
    for (int e = 0; e < 8; ++e) t8[e] = fmaxf(pv[e], pv[8 + e]);
#pragma unroll
    for (int e = 0; e < 4; ++e) t8[e] = fmaxf(t8[e], t8[4 + e]);
    float mx = fmaxf(fmaxf(t8[0], t8[1]), fmaxf(t8[2], t8[3]));
    mx = fmaxf(mx, xor16(mx));
    const float mnew = fmaxf(m_t, mx);
    const float alpha = __builtin_amdgcn_exp2f(m_t - mnew);
#pragma unroll
    for (int e = 0; e < 16; ++e)
      pv[e] = __builtin_amdgcn_exp2f(pv[e] - mnew);

    // branch-free causal mask after exp: zero p where token > row.
    // this lane's row = qbase + l16; token(kidx e) = j + half*8 + e (+16).
    // steady-state chunks: t2 >= 24 -> all selects no-ops.
    const int t2 = qbase + l16 - j - half * 8;
#pragma unroll
    for (int e = 0; e < 8; ++e) {
      if (e > t2) pv[e] = 0.0f;
      if (e + 16 > t2) pv[8 + e] = 0.0f;
    }
    float sum = 0.0f;
#pragma unroll
    for (int e = 0; e < 16; ++e) sum += pv[e];
    sum += xor16(sum);
    l_t = l_t * alpha + sum;
    m_t = mnew;

    // P fragment is already in A layout
    v16h pa;
#pragma unroll
    for (int e = 0; e < 16; ++e) pa[e] = (_Float16)pv[e];

    // broadcast alpha to C-layout rows (duplicate stores, 2 vector loads)
    Abuf[wv][l16] = alpha;
    float4 a0 = *(const float4*)&Abuf[wv][half * 8];
    float4 a1 = *(const float4*)&Abuf[wv][half * 8 + 4];
    float ac[8] = {a0.x, a0.y, a0.z, a0.w, a1.x, a1.y, a1.z, a1.w};
#pragma unroll
    for (int r = 0; r < 8; ++r)
#pragma unroll
      for (int nt2 = 0; nt2 < 4; ++nt2) o[nt2][r] *= ac[r];

    // ----- O += P * V  (4 hd tiles of 16), fragments already loaded -----
#pragma unroll
    for (int nt2 = 0; nt2 < 4; ++nt2)
      o[nt2] = __builtin_amdgcn_wmma_f32_16x16x32_f16(
          false, pa, false, vfr[nt2], (short)0, o[nt2], false, false);
  };

  const int maxch = qt * 4 + 4;                               // block-uniform
  const int myci  = __builtin_amdgcn_readfirstlane(qbase >> 5);  // wave last
  issue(0, 0);
  for (int ci = 0; ci < maxch; ++ci) {
    const int p = ci & 1;
    wait_async0();       // this wave's share of chunk ci has landed
    __syncthreads();     // everyone's share landed; prev-buffer readers done
    if (ci + 1 < maxch) issue(ci + 1, 1 - p);
    if (ci <= myci) process(ci * 32, p);
  }

  // ----- epilogue: normalize (broadcast 1/l) and store y [B][T][H*HD] -----
  const float linv = 1.0f / l_t;
  Abuf[wv][l16] = linv;
  float4 c0 = *(const float4*)&Abuf[wv][half * 8];
  float4 c1 = *(const float4*)&Abuf[wv][half * 8 + 4];
  float lc[8] = {c0.x, c0.y, c0.z, c0.w, c1.x, c1.y, c1.z, c1.w};
#pragma unroll
  for (int r = 0; r < 8; ++r) {
    int row = qbase + r + half * 8;
#pragma unroll
    for (int nt2 = 0; nt2 < 4; ++nt2) {
      yb[((size_t)(b * TT) + row) * DD + h * HD + nt2 * 16 + l16] =
          (_Float16)(o[nt2][r] * lc[r]);
    }
  }
}

// ---------------------------------------------------------------------------
extern "C" void kernel_launch(void* const* d_in, const int* in_sizes, int n_in,
                              void* d_out, int out_size, void* d_ws,
                              size_t ws_size, hipStream_t stream) {
  (void)in_sizes; (void)n_in; (void)out_size; (void)ws_size;
  const float* x     = (const float*)d_in[0];
  const float* Wq    = (const float*)d_in[1];
  const float* Wk    = (const float*)d_in[2];
  const float* Wv    = (const float*)d_in[3];
  const float* Wproj = (const float*)d_in[4];
  const float* qgain = (const float*)d_in[5];

  char* ws = (char*)d_ws;
  size_t off = 0;
  auto alloc = [&](size_t bytes) {
    void* p = ws + off;
    off += (bytes + 255) & ~(size_t)255;
    return p;
  };
  float*     qf = (float*)alloc((size_t)MTOT * DD * 4);          // 32 MB
  float*     kf = (float*)alloc((size_t)MTOT * HKV * HD * 4);    //  8 MB
  float*     vf = (float*)alloc((size_t)MTOT * HKV * HD * 4);    //  8 MB
  _Float16*  qh = (_Float16*)alloc((size_t)MTOT * DD * 2);       // 16 MB
  _Float16*  kh = (_Float16*)alloc((size_t)MTOT * HKV * HD * 2); //  4 MB
  _Float16*  vt = (_Float16*)alloc((size_t)MTOT * HKV * HD * 2); //  4 MB
  _Float16*  yb = (_Float16*)alloc((size_t)MTOT * DD * 2);       // 16 MB

  // 1) QKV projections (WMMA GEMMs, double-buffered LDS)
  gemm_xwt<float, 4><<<dim3(MTOT / 64, DD / 256), 256, 0, stream>>>(
      x, Wq, qf, MTOT, DD, DD);
  gemm_xwt<float, 2><<<dim3(MTOT / 64, (HKV * HD) / 128), 256, 0, stream>>>(
      x, Wk, kf, MTOT, HKV * HD, DD);
  gemm_xwt<float, 2><<<dim3(MTOT / 64, (HKV * HD) / 128), 256, 0, stream>>>(
      x, Wv, vf, MTOT, HKV * HD, DD);

  // 2) RMSNorm + RoPE (+gain/scale/log2e) -> f16 head-major
  normrope_kernel<<<(BB * TT * (HH + HKV)) / 8, 256, 0, stream>>>(
      qf, kf, qh, kh, qgain);

  // 3) V cast + tile-major transpose
  vtrans_kernel<<<((size_t)BB * HKV * HD * TT) / 256, 256, 0, stream>>>(vf, vt);

  // 4) causal flash attention (WMMA + block-shared async-to-LDS tiles)
  attn_kernel<<<BB * HH * (TT / 128), 256, 0, stream>>>(qh, kh, vt, yb);

  // 5) output projection (WMMA GEMM, f16 A)
  gemm_xwt<_Float16, 4><<<dim3(MTOT / 64, DD / 256), 256, 0, stream>>>(
      yb, Wproj, (float*)d_out, MTOT, DD, DD);
}
